// MultiHeadAttention_9895604650663
// MI455X (gfx1250) — compile-verified, matrix operand order checked by reading, not run on hardware
//
#include <hip/hip_runtime.h>

typedef __attribute__((ext_vector_type(16))) _Float16 v16h;
typedef __attribute__((ext_vector_type(8)))  _Float16 v8h;
typedef __attribute__((ext_vector_type(8)))  float    v8f;
typedef __attribute__((ext_vector_type(4)))  float    v4f;

#define D_MODEL   1024
#define NUM_HEADS 16
#define DEPTH     64
#define SEQ       2048
#define BATCH     2
#define MTOK      (BATCH * SEQ)   // 4096 tokens

// ---------------------------------------------------------------------------
// Fragment loaders. CDNA5 16-bit A/B layout (ISA 7.12.2): lane L holds row
// (L & 15); K elements are kh*8 + {0..7} in VGPRs 0..3 and 16 + kh*8 + {0..7}
// in VGPRs 4..7 (kh = L >> 4). Row-major source with the fragment's "row"
// dimension outermost => two contiguous 16B loads per lane.
// ---------------------------------------------------------------------------
__device__ __forceinline__ v16h load_frag(const _Float16* __restrict__ base,
                                          int ld, int row0, int col0) {
  const int lane = threadIdx.x & 31;
  const int m = lane & 15;
  const int kh = lane >> 4;
  const _Float16* p = base + (size_t)(row0 + m) * ld + col0 + kh * 8;
  v8h lo = *(const v8h*)p;
  v8h hi = *(const v8h*)(p + 16);
  v16h r;
#pragma unroll
  for (int i = 0; i < 8; ++i) { r[i] = lo[i]; r[i + 8] = hi[i]; }
  return r;
}

// Same layout, but source is fp32: convert to f16 on the fly.
__device__ __forceinline__ v16h load_frag(const float* __restrict__ base,
                                          int ld, int row0, int col0) {
  const int lane = threadIdx.x & 31;
  const int m = lane & 15;
  const int kh = lane >> 4;
  const float* p = base + (size_t)(row0 + m) * ld + col0 + kh * 8;
  v4f a0 = *(const v4f*)p;
  v4f a1 = *(const v4f*)(p + 4);
  v4f b0 = *(const v4f*)(p + 16);
  v4f b1 = *(const v4f*)(p + 20);
  v16h r;
#pragma unroll
  for (int i = 0; i < 4; ++i) {
    r[i]      = (_Float16)a0[i];
    r[i + 4]  = (_Float16)a1[i];
    r[i + 8]  = (_Float16)b0[i];
    r[i + 12] = (_Float16)b1[i];
  }
  return r;
}

// ---------------------------------------------------------------------------
// Weight convert + transpose: W fp32 [in, out] -> Wt f16 [out, in].
// LDS-tiled so both global reads and writes are coalesced.
// ---------------------------------------------------------------------------
__global__ __launch_bounds__(256) void transpose_w(const float* __restrict__ W,
                                                   _Float16* __restrict__ Wt) {
  __shared__ float tile[32][33];
  const int tx = threadIdx.x & 31;
  const int ty = threadIdx.x >> 5;          // 0..7
  const int i0 = blockIdx.y * 32;           // input row block  (in  dim)
  const int o0 = blockIdx.x * 32;           // input col block  (out dim)
#pragma unroll
  for (int j = 0; j < 32; j += 8)
    tile[ty + j][tx] = W[(size_t)(i0 + ty + j) * D_MODEL + o0 + tx];
  __syncthreads();
#pragma unroll
  for (int j = 0; j < 32; j += 8)
    Wt[(size_t)(o0 + ty + j) * D_MODEL + i0 + tx] = (_Float16)tile[tx][ty + j];
}

// ---------------------------------------------------------------------------
// GEMM: Out[M, 1024] = A[M, 1024] @ Bt^T + bias    (Bt is [N, K] row-major)
// Block = 256 threads = 8 waves; block tile 128x128; wave tile 64x32.
// MODE 0: store f16 row-major (Q, K)
// MODE 1: store f16 transposed per head -> Vt[b][h][d][s] (V)
// MODE 2: store f32 row-major (final output)
// ---------------------------------------------------------------------------
template <typename AT, int MODE>
__global__ __launch_bounds__(256) void gemm_wmma(const AT* __restrict__ A,
                                                 const _Float16* __restrict__ Bt,
                                                 const float* __restrict__ bias,
                                                 void* __restrict__ Out) {
  const int warp = threadIdx.x >> 5;
  const int lane = threadIdx.x & 31;
  const int wy = warp >> 2;
  const int wx = warp & 3;
  const int M0 = blockIdx.y * 128 + wy * 64;
  const int N0 = blockIdx.x * 128 + wx * 32;

  v8f acc[4][2];
#pragma unroll
  for (int mi = 0; mi < 4; ++mi)
#pragma unroll
    for (int ni = 0; ni < 2; ++ni) acc[mi][ni] = {};

  for (int k0 = 0; k0 < D_MODEL; k0 += 32) {
    v16h a[4], b[2];
#pragma unroll
    for (int mi = 0; mi < 4; ++mi) a[mi] = load_frag(A, D_MODEL, M0 + mi * 16, k0);
#pragma unroll
    for (int ni = 0; ni < 2; ++ni) b[ni] = load_frag(Bt, D_MODEL, N0 + ni * 16, k0);
#pragma unroll
    for (int mi = 0; mi < 4; ++mi)
#pragma unroll
      for (int ni = 0; ni < 2; ++ni)
        acc[mi][ni] = __builtin_amdgcn_wmma_f32_16x16x32_f16(
            false, a[mi], false, b[ni], (short)0, acc[mi][ni], false, false);
  }

  const int kh = lane >> 4;
  const int ln = lane & 15;
#pragma unroll
  for (int ni = 0; ni < 2; ++ni) {
    const int col = N0 + ni * 16 + ln;
    const float bv = bias[col];
#pragma unroll
    for (int mi = 0; mi < 4; ++mi) {
#pragma unroll
      for (int r = 0; r < 8; ++r) {
        const int row = M0 + mi * 16 + r + 8 * kh;   // C layout: VGPR r, half kh
        const float val = acc[mi][ni][r] + bv;
        if (MODE == 0) {
          ((_Float16*)Out)[(size_t)row * D_MODEL + col] = (_Float16)val;
        } else if (MODE == 1) {
          const int h = col >> 6, d = col & 63;
          const int bb = row >> 11, s = row & (SEQ - 1);
          ((_Float16*)Out)[(((size_t)bb * NUM_HEADS + h) * DEPTH + d) * SEQ + s] =
              (_Float16)val;
        } else {
          ((float*)Out)[(size_t)row * D_MODEL + col] = val;
        }
      }
    }
  }
}

// ---------------------------------------------------------------------------
// Flash attention: 1 wave per 32-query tile, stream KV in chunks of 64 with
// online softmax (wide chunk halves shuffle/rescale overhead per key);
// P chunk staged through LDS to convert C-layout -> A-layout.
// Block = 128 threads = 4 waves (4 independent q tiles, same loop counts).
// ---------------------------------------------------------------------------
__global__ __launch_bounds__(128) void attn_kernel(const _Float16* __restrict__ Qf,
                                                   const _Float16* __restrict__ Kf,
                                                   const _Float16* __restrict__ Vt,
                                                   _Float16* __restrict__ Ctx) {
  __shared__ _Float16 smem[4][32][64];   // per-wave 32x64 f16 P staging (16 KB)
  const int warp = threadIdx.x >> 5;
  const int lane = threadIdx.x & 31;
  const int t  = blockIdx.x * 4 + warp;          // global q-tile id (0..2047)
  const int qt = t & 63;                          // 64 q-tiles per (b,h)
  const int h  = (t >> 6) & (NUM_HEADS - 1);
  const int b  = t >> 10;
  const int q0 = qt * 32;
  const int tok0 = b * SEQ + q0;
  const int kh = lane >> 4;
  const int ln = lane & 15;

  // Q fragments resident in registers: 2 (m) x 2 (k of depth-64) tiles.
  v16h qf[2][2];
#pragma unroll
  for (int mi = 0; mi < 2; ++mi)
#pragma unroll
    for (int kk = 0; kk < 2; ++kk)
      qf[mi][kk] = load_frag(Qf, D_MODEL, tok0 + mi * 16, h * DEPTH + kk * 32);

  v8f o[2][4];
#pragma unroll
  for (int mi = 0; mi < 2; ++mi)
#pragma unroll
    for (int nd = 0; nd < 4; ++nd) o[mi][nd] = {};

  // Per-lane running max/sum for the 16 rows this lane's C-half owns.
  float mrun[16], lrun[16];
#pragma unroll
  for (int i = 0; i < 16; ++i) { mrun[i] = -1e30f; lrun[i] = 0.0f; }

  const _Float16* vtb = Vt + ((size_t)(b * NUM_HEADS + h) * DEPTH) * SEQ;
  const float scale = 0.125f;   // 1/sqrt(DEPTH)

  for (int kb = 0; kb < SEQ; kb += 64) {
    // ---- S = Q @ K^T  (32 q x 64 kv) --------------------------------------
    v8f s[2][4];
#pragma unroll
    for (int mi = 0; mi < 2; ++mi)
#pragma unroll
      for (int ni = 0; ni < 4; ++ni) s[mi][ni] = {};

#pragma unroll
    for (int kk = 0; kk < 2; ++kk) {
#pragma unroll
      for (int ni = 0; ni < 4; ++ni) {
        v16h bf = load_frag(Kf, D_MODEL, b * SEQ + kb + ni * 16, h * DEPTH + kk * 32);
#pragma unroll
        for (int mi = 0; mi < 2; ++mi)
          s[mi][ni] = __builtin_amdgcn_wmma_f32_16x16x32_f16(
              false, qf[mi][kk], false, bf, (short)0, s[mi][ni], false, false);
      }
    }

    // ---- online softmax (row reductions inside 16-lane halves) ------------
#pragma unroll
    for (int mi = 0; mi < 2; ++mi) {
#pragma unroll
      for (int r = 0; r < 8; ++r) {
        const int idx = mi * 8 + r;
        float sv[4];
#pragma unroll
        for (int ni = 0; ni < 4; ++ni) sv[ni] = s[mi][ni][r] * scale;
        float cm = fmaxf(fmaxf(sv[0], sv[1]), fmaxf(sv[2], sv[3]));
#pragma unroll
        for (int off = 8; off >= 1; off >>= 1) cm = fmaxf(cm, __shfl_xor(cm, off, 32));
        const float nm = fmaxf(mrun[idx], cm);
        const float al = __expf(mrun[idx] - nm);
        float p[4];
        float rs = 0.0f;
#pragma unroll
        for (int ni = 0; ni < 4; ++ni) { p[ni] = __expf(sv[ni] - nm); rs += p[ni]; }
#pragma unroll
        for (int off = 8; off >= 1; off >>= 1) rs += __shfl_xor(rs, off, 32);
        lrun[idx] = lrun[idx] * al + rs;
        mrun[idx] = nm;
#pragma unroll
        for (int nd = 0; nd < 4; ++nd) o[mi][nd][r] *= al;
#pragma unroll
        for (int ni = 0; ni < 4; ++ni) s[mi][ni][r] = p[ni];
      }
    }

    // ---- P: C-layout -> LDS -> A-layout -----------------------------------
#pragma unroll
    for (int mi = 0; mi < 2; ++mi)
#pragma unroll
      for (int ni = 0; ni < 4; ++ni)
#pragma unroll
        for (int r = 0; r < 8; ++r)
          smem[warp][mi * 16 + r + 8 * kh][ni * 16 + ln] = (_Float16)s[mi][ni][r];
    __syncthreads();

    v16h pa[2][2];
#pragma unroll
    for (int mi = 0; mi < 2; ++mi)
#pragma unroll
      for (int kk2 = 0; kk2 < 2; ++kk2)
        pa[mi][kk2] = load_frag(&smem[warp][mi * 16][0], 64, 0, kk2 * 32);
    __syncthreads();

    // ---- O += P @ V  (Vt rows = d, contiguous over keys) ------------------
#pragma unroll
    for (int nd = 0; nd < 4; ++nd) {
#pragma unroll
      for (int kk2 = 0; kk2 < 2; ++kk2) {
        v16h vf = load_frag(vtb, SEQ, nd * 16, kb + kk2 * 32);
#pragma unroll
        for (int mi = 0; mi < 2; ++mi)
          o[mi][nd] = __builtin_amdgcn_wmma_f32_16x16x32_f16(
              false, pa[mi][kk2], false, vf, (short)0, o[mi][nd], false, false);
      }
    }
  }

  // ---- normalize & write context (f16 row-major [token, d_model]) ---------
#pragma unroll
  for (int mi = 0; mi < 2; ++mi) {
#pragma unroll
    for (int r = 0; r < 8; ++r) {
      const float inv = 1.0f / lrun[mi * 8 + r];
      const int row = tok0 + mi * 16 + r + 8 * kh;
#pragma unroll
      for (int nd = 0; nd < 4; ++nd) {
        const int col = h * DEPTH + nd * 16 + ln;
        Ctx[(size_t)row * D_MODEL + col] = (_Float16)(o[mi][nd][r] * inv);
      }
    }
  }
}

// ---------------------------------------------------------------------------
extern "C" void kernel_launch(void* const* d_in, const int* in_sizes, int n_in,
                              void* d_out, int out_size, void* d_ws, size_t ws_size,
                              hipStream_t stream) {
  (void)in_sizes; (void)n_in; (void)out_size; (void)ws_size;
  const float* q  = (const float*)d_in[0];
  const float* k  = (const float*)d_in[1];
  const float* v  = (const float*)d_in[2];
  const float* Wq = (const float*)d_in[3];
  const float* bq = (const float*)d_in[4];
  const float* Wk = (const float*)d_in[5];
  const float* bk = (const float*)d_in[6];
  const float* Wv = (const float*)d_in[7];
  const float* bv = (const float*)d_in[8];
  const float* Wo = (const float*)d_in[9];
  const float* bo = (const float*)d_in[10];

  char* ws = (char*)d_ws;
  const size_t WB = (size_t)D_MODEL * D_MODEL * sizeof(_Float16);  // 2 MB
  const size_t AB = (size_t)MTOK * D_MODEL * sizeof(_Float16);     // 8 MB
  _Float16* WqT = (_Float16*)(ws + 0 * WB);
  _Float16* WkT = (_Float16*)(ws + 1 * WB);
  _Float16* WvT = (_Float16*)(ws + 2 * WB);
  _Float16* WoT = (_Float16*)(ws + 3 * WB);
  _Float16* Qf  = (_Float16*)(ws + 4 * WB + 0 * AB);
  _Float16* Kf  = (_Float16*)(ws + 4 * WB + 1 * AB);
  _Float16* Vtp = (_Float16*)(ws + 4 * WB + 2 * AB);
  _Float16* Ctx = (_Float16*)(ws + 4 * WB + 3 * AB);   // total 40 MB

  dim3 tg(D_MODEL / 32, D_MODEL / 32);                 // 32 x 32 blocks
  transpose_w<<<tg, 256, 0, stream>>>(Wq, WqT);
  transpose_w<<<tg, 256, 0, stream>>>(Wk, WkT);
  transpose_w<<<tg, 256, 0, stream>>>(Wv, WvT);
  transpose_w<<<tg, 256, 0, stream>>>(Wo, WoT);

  dim3 gg(D_MODEL / 128, MTOK / 128);                  // 8 x 32 blocks
  gemm_wmma<float, 0><<<gg, 256, 0, stream>>>(q, WqT, bq, Qf);
  gemm_wmma<float, 0><<<gg, 256, 0, stream>>>(k, WkT, bk, Kf);
  gemm_wmma<float, 1><<<gg, 256, 0, stream>>>(v, WvT, bv, Vtp);

  attn_kernel<<<512, 128, 0, stream>>>(Qf, Kf, Vtp, Ctx);

  gemm_wmma<_Float16, 2><<<gg, 256, 0, stream>>>(Ctx, WoT, bo, d_out);
}
